// vRNNLayer_4183298146629
// MI455X (gfx1250) — matrix operand order repaired and checked by reading, not compiled, not run on hardware
//
#include <hip/hip_runtime.h>

// Problem constants (match reference)
#define T_STEPS 512
#define BATCH   64
#define IN_F    512
#define HID_F   2048
#define OUT_F   512
#define ALPHA_F 0.1f

typedef __bf16 bf16_t;
typedef __attribute__((ext_vector_type(16))) __bf16 v16bf;
typedef __attribute__((ext_vector_type(8)))  float  v8f;

union FragU { v16bf v; uint4 q[2]; };

// ---------------------------------------------------------------------------
// Fragment loaders.
// A (16x32 bf16): lane holds row M=lane&15; K elements are two 16B chunks at
//   kb..kb+7 and kb+16..kb+23 where kb = k0 + 8*(lane>=16).   (ISA 7.12.2)
// B (32x16 bf16): lane holds column N=lane&15; K elements are two 16B chunks at
//   kb..kb+7 and kb+16..kb+23 where kb = k0 + 8*(lane>=16).
// ---------------------------------------------------------------------------
__device__ __forceinline__ v16bf load_frag_a(const bf16_t* __restrict__ rowp,
                                             int k0, int lane) {
    const int kb = k0 + ((lane >> 4) << 3);
    FragU t;
    t.q[0] = *(const uint4*)(rowp + kb);
    t.q[1] = *(const uint4*)(rowp + kb + 16);
    return t.v;
}

__device__ __forceinline__ v16bf load_frag_b(const bf16_t* __restrict__ colp,
                                             int k0, int lane) {
    const int kb = k0 + ((lane >> 4) << 3);
    FragU t;
    t.q[0] = *(const uint4*)(colp + kb);
    t.q[1] = *(const uint4*)(colp + kb + 16);
    return t.v;
}

// ---------------------------------------------------------------------------
// f32 -> bf16 conversion, 4 elements / thread (8B stores)
// ---------------------------------------------------------------------------
__global__ void cvt_f32_bf16_kernel(const float* __restrict__ s,
                                    bf16_t* __restrict__ d, int n4) {
    int i = blockIdx.x * blockDim.x + threadIdx.x;
    if (i < n4) {
        float4 f = ((const float4*)s)[i];
        union { bf16_t b[4]; uint2 raw; } t;
        t.b[0] = (bf16_t)f.x; t.b[1] = (bf16_t)f.y;
        t.b[2] = (bf16_t)f.z; t.b[3] = (bf16_t)f.w;
        ((uint2*)d)[i] = t.raw;
    }
}

// ---------------------------------------------------------------------------
// Input projection: U[T*B, HID] = bf16( X[T*B, IN] @ Win[HID, IN]^T + b_in )
// One wave -> 16 rows x 64 cols, K = IN_F. Software-pipelined fragment loads.
// ---------------------------------------------------------------------------
__global__ void uproj_kernel(const bf16_t* __restrict__ X,
                             const bf16_t* __restrict__ Win,
                             const float*  __restrict__ bin,
                             bf16_t* __restrict__ U) {
    const int lane = threadIdx.x & 31;
    const int wid  = threadIdx.x >> 5;
    const int w    = blockIdx.x * 8 + wid;

    const int m_tile = w >> 5;          // 0 .. 2047
    const int g      = w & 31;          // 0 .. 31
    const int row0   = m_tile * 16;
    const int n0     = g * 64;

    const int lrow = lane & 15;
    const int hi8  = (lane >> 4) << 3;

    v8f acc[4];
    const v8f vzero = {0.f,0.f,0.f,0.f,0.f,0.f,0.f,0.f};
#pragma unroll
    for (int s = 0; s < 4; ++s) acc[s] = vzero;

    const bf16_t* arow = X + (size_t)(row0 + lrow) * IN_F;
    const bf16_t* bcol[4];
#pragma unroll
    for (int s = 0; s < 4; ++s)
        bcol[s] = Win + (size_t)(n0 + s * 16 + lrow) * IN_F;

    // ---- software pipeline: prime stage 0 ----
    v16bf a = load_frag_a(arow, 0, lane);
    v16bf b[4];
#pragma unroll
    for (int s = 0; s < 4; ++s) b[s] = load_frag_b(bcol[s], 0, lane);

    for (int k0 = 0; k0 < IN_F - 32; k0 += 32) {
        const int kn = k0 + 32;
        v16bf a_n = load_frag_a(arow, kn, lane);
        v16bf b_n[4];
#pragma unroll
        for (int s = 0; s < 4; ++s) b_n[s] = load_frag_b(bcol[s], kn, lane);
#pragma unroll
        for (int s = 0; s < 4; ++s)
            acc[s] = __builtin_amdgcn_wmma_f32_16x16x32_bf16(
                false, a, false, b[s], (short)0, acc[s], false, false);
        a = a_n;
#pragma unroll
        for (int s = 0; s < 4; ++s) b[s] = b_n[s];
    }
#pragma unroll
    for (int s = 0; s < 4; ++s)         // peeled last stage
        acc[s] = __builtin_amdgcn_wmma_f32_16x16x32_bf16(
            false, a, false, b[s], (short)0, acc[s], false, false);

    // Epilogue: C layout -> lane holds column (n), VGPR r holds row r (+8 hi half)
#pragma unroll
    for (int s = 0; s < 4; ++s) {
        const int col = n0 + s * 16 + lrow;
        const float bias = bin[col];
#pragma unroll
        for (int r = 0; r < 8; ++r) {
            const int row = row0 + r + hi8;
            U[(size_t)row * HID_F + col] = (bf16_t)(acc[s][r] + bias);
        }
    }
}

// ---------------------------------------------------------------------------
// One recurrence step (fused recurrent GEMM + readout GEMM + update).
//   g in [0,32):  pre = h@Whh^T ; h_next = h + a*(-h + tanh(pre + b_hh + u_t))
//   g in [32,40): out_t = h@Wout^T + b_out          (readout uses PRE-update h)
// 160 waves = 40 blocks x 4 waves; the 4 waves of a block are the 4 M-tiles of
// ONE N-group, so they share all B (weight) fragments in L0.
// ---------------------------------------------------------------------------
__global__ void rnn_step_kernel(const bf16_t* __restrict__ h_in_bf,
                                const float*  __restrict__ h_in_f32,
                                bf16_t* __restrict__ h_out_bf,
                                float*  __restrict__ h_out_f32,
                                const bf16_t* __restrict__ Whh,
                                const float*  __restrict__ bhh,
                                const bf16_t* __restrict__ Wout,
                                const float*  __restrict__ bout,
                                const bf16_t* __restrict__ u_t,
                                float* __restrict__ out_t) {
    const int lane = threadIdx.x & 31;
    const int wid  = threadIdx.x >> 5;
    const int w    = blockIdx.x * 4 + wid;

    const int m_tile = w & 3;           // 4 M tiles cover B = 64
    const int g      = w >> 2;          // 0 .. 39 N groups of 64 (same g per block)
    const int m0     = m_tile * 16;

    const int lrow = lane & 15;
    const int hi8  = (lane >> 4) << 3;

    const bf16_t* Bp;
    int ncol;
    if (g < 32) { Bp = Whh;  ncol = g * 64; }
    else        { Bp = Wout; ncol = (g - 32) * 64; }

    v8f acc[4];
    const v8f vzero = {0.f,0.f,0.f,0.f,0.f,0.f,0.f,0.f};
#pragma unroll
    for (int s = 0; s < 4; ++s) acc[s] = vzero;

    const bf16_t* arow = h_in_bf + (size_t)(m0 + lrow) * HID_F;
    const bf16_t* bcol[4];
#pragma unroll
    for (int s = 0; s < 4; ++s)
        bcol[s] = Bp + (size_t)(ncol + s * 16 + lrow) * HID_F;

    // ---- software pipeline: prime stage 0 ----
    v16bf a = load_frag_a(arow, 0, lane);
    v16bf b[4];
#pragma unroll
    for (int s = 0; s < 4; ++s) b[s] = load_frag_b(bcol[s], 0, lane);

    for (int k0 = 0; k0 < HID_F - 32; k0 += 32) {
        const int kn = k0 + 32;
        v16bf a_n = load_frag_a(arow, kn, lane);
        v16bf b_n[4];
#pragma unroll
        for (int s = 0; s < 4; ++s) b_n[s] = load_frag_b(bcol[s], kn, lane);
#pragma unroll
        for (int s = 0; s < 4; ++s)
            acc[s] = __builtin_amdgcn_wmma_f32_16x16x32_bf16(
                false, a, false, b[s], (short)0, acc[s], false, false);
        a = a_n;
#pragma unroll
        for (int s = 0; s < 4; ++s) b[s] = b_n[s];
    }
#pragma unroll
    for (int s = 0; s < 4; ++s)         // peeled last stage
        acc[s] = __builtin_amdgcn_wmma_f32_16x16x32_bf16(
            false, a, false, b[s], (short)0, acc[s], false, false);

    if (g < 32) {
        // recurrent path: leak update on hidden state, write both precisions
#pragma unroll
        for (int s = 0; s < 4; ++s) {
            const int col = ncol + s * 16 + lrow;
            const float bias = bhh[col];
#pragma unroll
            for (int r = 0; r < 8; ++r) {
                const int row = m0 + r + hi8;
                const size_t idx = (size_t)row * HID_F + col;
                const float pre = acc[s][r] + bias + (float)u_t[idx];
                const float hv  = h_in_f32[idx];
                const float hn  = hv + ALPHA_F * (tanhf(pre) - hv);
                h_out_f32[idx] = hn;
                h_out_bf[idx]  = (bf16_t)hn;
            }
        }
    } else {
        // readout path
#pragma unroll
        for (int s = 0; s < 4; ++s) {
            const int col = ncol + s * 16 + lrow;
            const float bias = bout[col];
#pragma unroll
            for (int r = 0; r < 8; ++r) {
                const int row = m0 + r + hi8;
                out_t[(size_t)row * OUT_F + col] = acc[s][r] + bias;
            }
        }
    }
}

// ---------------------------------------------------------------------------
// Host side
// ---------------------------------------------------------------------------
extern "C" void kernel_launch(void* const* d_in, const int* in_sizes, int n_in,
                              void* d_out, int out_size, void* d_ws, size_t ws_size,
                              hipStream_t stream) {
    const float* input = (const float*)d_in[0];   // [T,B,IN]
    const float* Win   = (const float*)d_in[1];   // [HID,IN]
    const float* bin   = (const float*)d_in[2];   // [HID]
    const float* Whh   = (const float*)d_in[3];   // [HID,HID]
    const float* bhh   = (const float*)d_in[4];   // [HID]
    const float* Wout  = (const float*)d_in[5];   // [OUT,HID]
    const float* bout  = (const float*)d_in[6];   // [OUT]
    float* out = (float*)d_out;                   // [T,B,OUT]

    char* ws = (char*)d_ws;
    size_t off = 0;
    bf16_t* Xb    = (bf16_t*)(ws + off); off += (size_t)T_STEPS * BATCH * IN_F  * 2;
    bf16_t* Winb  = (bf16_t*)(ws + off); off += (size_t)HID_F  * IN_F          * 2;
    bf16_t* Whhb  = (bf16_t*)(ws + off); off += (size_t)HID_F  * HID_F         * 2;
    bf16_t* Woutb = (bf16_t*)(ws + off); off += (size_t)OUT_F  * HID_F         * 2;
    bf16_t* Ub    = (bf16_t*)(ws + off); off += (size_t)T_STEPS * BATCH * HID_F * 2;
    float*  h32   = (float*) (ws + off); off += (size_t)2 * BATCH * HID_F * 4;
    bf16_t* h16   = (bf16_t*)(ws + off); off += (size_t)2 * BATCH * HID_F * 2;

    // 1) bf16 conversions (all counts divisible by 4)
    {
        int n4;
        n4 = T_STEPS * BATCH * IN_F / 4;
        cvt_f32_bf16_kernel<<<(n4 + 255) / 256, 256, 0, stream>>>(input, Xb, n4);
        n4 = HID_F * IN_F / 4;
        cvt_f32_bf16_kernel<<<(n4 + 255) / 256, 256, 0, stream>>>(Win, Winb, n4);
        n4 = HID_F * HID_F / 4;
        cvt_f32_bf16_kernel<<<(n4 + 255) / 256, 256, 0, stream>>>(Whh, Whhb, n4);
        n4 = OUT_F * HID_F / 4;
        cvt_f32_bf16_kernel<<<(n4 + 255) / 256, 256, 0, stream>>>(Wout, Woutb, n4);
    }

    // 2) h0 = 0 (ping buffer only; pong is fully written before first read)
    hipMemsetAsync(h32, 0, (size_t)BATCH * HID_F * 4, stream);
    hipMemsetAsync(h16, 0, (size_t)BATCH * HID_F * 2, stream);

    // 3) input projection U = X @ Win^T + b_in  (saturating GEMM)
    {
        const int waves  = (T_STEPS * BATCH / 16) * (HID_F / 64); // 65536
        const int blocks = waves / 8;                              // 8192
        uproj_kernel<<<blocks, 256, 0, stream>>>(Xb, Winb, bin, Ub);
    }

    // 4) 512 sequential steps, ping-pong hidden state
    const size_t hElems = (size_t)BATCH * HID_F;
    for (int t = 0; t < T_STEPS; ++t) {
        const int cur = t & 1, nxt = cur ^ 1;
        rnn_step_kernel<<<40, 128, 0, stream>>>(
            h16 + (size_t)cur * hElems, h32 + (size_t)cur * hElems,
            h16 + (size_t)nxt * hElems, h32 + (size_t)nxt * hElems,
            Whhb, bhh, Woutb, bout,
            Ub  + (size_t)t * hElems,
            out + (size_t)t * BATCH * OUT_F);
    }
}
